// Reservoir_61813169324611
// MI455X (gfx1250) — matrix operand order repaired and checked by reading, not compiled, not run on hardware
//
#include <hip/hip_runtime.h>
#include <hip/hip_bf16.h>

typedef __attribute__((ext_vector_type(16))) _Float16 v16h;
typedef __attribute__((ext_vector_type(8)))  _Float16 v8h;
typedef __attribute__((ext_vector_type(8)))  float    v8f;

#define T_STEPS 15
#define CIN     6
#define HIN     96
#define WIN     96
#define NF      512
#define HO      92
#define WO      92
#define HWO     (HO*WO)            // 8464
#define MROWS   (T_STEPS*HWO)      // 126960
#define KCONV   150
#define KCONVP  160
#define THRESH  15.0f

// ---------------------------------------------------------------- prep: f32->f16 weights
__global__ __launch_bounds__(256) void prep_weights(
    const float* __restrict__ wconv, const float* __restrict__ wrec,
    _Float16* __restrict__ wconv_h, _Float16* __restrict__ wrec_h)
{
  int i = blockIdx.x * 256 + threadIdx.x;
  if (i < NF * KCONVP) {
    int f = i / KCONVP, k = i % KCONVP;
    wconv_h[i] = (k < KCONV) ? (_Float16)wconv[f * KCONV + k] : (_Float16)0.f;
  }
  if (i < NF * NF) wrec_h[i] = (_Float16)wrec[i];
}

// ---------------------------------------------------------------- conv (implicit GEMM, WMMA) + fire1
__global__ __launch_bounds__(256) void conv_fire1(
    const float* __restrict__ inp,            // (T,CIN,96,96)
    const _Float16* __restrict__ wconv_h,     // (NF,160)
    float* __restrict__ pot_out,              // (T,NF,HWO)  pot1 staged here
    _Float16* __restrict__ spk1h)             // (MROWS,NF)
{
  __shared__ __align__(16) _Float16 Asm[16 * KCONVP];   // 5 KB im2col tile [m][k]
  const int tx = blockIdx.x;                  // 0..5 x-tiles
  const int y  = blockIdx.y;                  // 0..91
  const int t  = blockIdx.z;                  // 0..14
  const int x0 = (tx < 5) ? tx * 16 : (WO - 16);
  const int tid = threadIdx.x;

  // im2col into LDS: A[m][k] = in[t, c, y+ky, x0+m+kx], k=(c*5+ky)*5+kx, pad k>=150
  for (int idx = tid; idx < 16 * KCONVP; idx += 256) {
    int m = idx / KCONVP, k = idx % KCONVP;
    float v = 0.f;
    if (k < KCONV) {
      int c = k / 25, r = k % 25, ky = r / 5, kx = r % 5;
      v = inp[((t * CIN + c) * HIN + (y + ky)) * WIN + (x0 + m + kx)];
    }
    Asm[m * KCONVP + k] = (_Float16)v;
  }
  __syncthreads();

  const int lane = tid & 31;
  const int wave = tid >> 5;
  const int half = lane >> 4;       // 0: lanes 0-15, 1: lanes 16-31
  const int l16  = lane & 15;
  const int n0   = wave * 64;       // 8 waves x 64 = 512 features

  v8f acc[4] = {};
  for (int k0 = 0; k0 < KCONVP; k0 += 32) {
    // A fragment: lane<16 -> (M=l16, K k0+0..15); lane>=16 -> K k0+16..31
    const _Float16* ar = &Asm[l16 * KCONVP + k0 + half * 16];
    v8h alo = *(const v8h*)(ar);
    v8h ahi = *(const v8h*)(ar + 8);
    v16h a = __builtin_shufflevector(alo, ahi, 0,1,2,3,4,5,6,7,8,9,10,11,12,13,14,15);
#pragma unroll
    for (int tle = 0; tle < 4; ++tle) {
      int n = n0 + tle * 16 + l16;  // B column per lane: 16 contiguous K halves
      v16h b = *(const v16h*)(wconv_h + n * KCONVP + k0 + half * 16);
      acc[tle] = __builtin_amdgcn_wmma_f32_16x16x32_f16(
          false, a, false, b, (short)0, acc[tle], false, false);
    }
  }

  // epilogue: fire1; stage pot1 (f32) into pot_out, spk1 (f16 0/1) K-major for GEMM A
  const int hwbase = y * WO + x0;
#pragma unroll
  for (int tle = 0; tle < 4; ++tle) {
    int n = n0 + tle * 16 + l16;
#pragma unroll
    for (int r = 0; r < 8; ++r) {
      int m = r + half * 8;                   // C layout: VGPR r -> M=r (lo) / M=8+r (hi)
      float pv = acc[tle][r];
      bool fired = (pv >= THRESH);
      pot_out[(size_t)(t * NF + n) * HWO + hwbase + m] = fired ? pv : 0.f;
      spk1h[((size_t)(t * HWO + hwbase + m)) * NF + n] = fired ? (_Float16)1.f
                                                               : (_Float16)0.f;
    }
  }
}

// ---------------------------------------------------------------- recurrent GEMM (WMMA) + fire2
__global__ __launch_bounds__(256) void gemm_fire2(
    const _Float16* __restrict__ spk1h,       // (MROWS,NF)
    const _Float16* __restrict__ wrec_h,      // (NF,NF) row = output feature g
    float* __restrict__ pot_out,              // in: pot1, out: pot2
    float* __restrict__ spk_out)              // spk2 (f32 0/1)
{
  const int m0  = blockIdx.x * 16;            // 7935 blocks, exact
  const int tid = threadIdx.x;
  const int lane = tid & 31, wave = tid >> 5;
  const int half = lane >> 4, l16 = lane & 15;
  const int n0 = wave * 64;
  const int t   = m0 / HWO;                   // 8464 % 16 == 0 -> no t straddle
  const int hw0 = m0 % HWO;

  v8f acc[4] = {};
  const _Float16* arow = spk1h + (size_t)(m0 + l16) * NF + half * 16;
#pragma unroll 4
  for (int k0 = 0; k0 < NF; k0 += 32) {
    __builtin_prefetch((const void*)(arow + k0 + 128), 0, 1);
    v8h alo = *(const v8h*)(arow + k0);
    v8h ahi = *(const v8h*)(arow + k0 + 8);
    v16h a = __builtin_shufflevector(alo, ahi, 0,1,2,3,4,5,6,7,8,9,10,11,12,13,14,15);
#pragma unroll
    for (int tle = 0; tle < 4; ++tle) {
      int g = n0 + tle * 16 + l16;
      v16h b = *(const v16h*)(wrec_h + (size_t)g * NF + k0 + half * 16);
      acc[tle] = __builtin_amdgcn_wmma_f32_16x16x32_f16(
          false, a, false, b, (short)0, acc[tle], false, false);
    }
  }

#pragma unroll
  for (int tle = 0; tle < 4; ++tle) {
    int n = n0 + tle * 16 + l16;
    size_t base = (size_t)(t * NF + n) * HWO + hw0;
#pragma unroll
    for (int r = 0; r < 8; ++r) {
      int mloc = r + half * 8;
      float val = pot_out[base + mloc] + acc[tle][r];
      bool fired = (val >= THRESH);
      pot_out[base + mloc] = fired ? val : 0.f;
      spk_out[base + mloc] = fired ? 1.f : 0.f;
    }
  }
}

// ---------------------------------------------------------------- pointwise inhibition stats per (h,w)
__global__ __launch_bounds__(256) void inhibit_stats(
    const float* __restrict__ pot2,           // (T,NF,HWO)
    int* __restrict__ winmap, float* __restrict__ nmap,
    float* __restrict__ valmap, float* __restrict__ cand)
{
  __shared__ float rv[256];
  __shared__ int   ri[256];
  __shared__ float tmax[T_STEPS];
  __shared__ int   targ[T_STEPS];
  const int hw  = blockIdx.x;
  const int tid = threadIdx.x;

  for (int t = 0; t < T_STEPS; ++t) {
    float bv = -1.f; int bi = 0;
    for (int f = tid; f < NF; f += 256) {
      float v = pot2[(size_t)(t * NF + f) * HWO + hw];
      if (v > bv) { bv = v; bi = f; }         // argmax: first (lowest f) on ties
    }
    rv[tid] = bv; ri[tid] = bi;
    __syncthreads();
    for (int s = 128; s > 0; s >>= 1) {
      if (tid < s) {
        float ov = rv[tid + s]; int oi = ri[tid + s];
        if (ov > rv[tid] || (ov == rv[tid] && oi < ri[tid])) { rv[tid] = ov; ri[tid] = oi; }
      }
      __syncthreads();
    }
    if (tid == 0) { tmax[t] = rv[0]; targ[t] = ri[0]; }
    __syncthreads();
  }

  if (tid == 0) {
    int nspk = 0;
    for (int t = 0; t < T_STEPS; ++t) nspk += (tmax[t] > 0.f) ? 1 : 0;
    int tf = T_STEPS - nspk; tf = (tf < 0) ? 0 : ((tf > T_STEPS - 1) ? T_STEPS - 1 : tf);
    int winner = targ[tf];
    bool has = tmax[T_STEPS - 1] > 0.f;       // any spike by last timestep
    int n = 0;
    if (has)
      for (int t = 0; t < T_STEPS; ++t)
        n += (pot2[(size_t)(t * NF + winner) * HWO + hw] > 0.f) ? 1 : 0;
    int ft2 = T_STEPS - n; ft2 = (ft2 < 0) ? 0 : ((ft2 > T_STEPS - 1) ? T_STEPS - 1 : ft2);
    float value = has ? pot2[(size_t)(ft2 * NF + winner) * HWO + hw] : 0.f;
    winmap[hw] = winner;
    nmap[hw]   = (float)n;
    valmap[hw] = value;
    cand[hw]   = (n > 0) ? value : 0.f;       // candidate for global max(trunc)
  }
}

// ---------------------------------------------------------------- total = n*(value + max(trunc)*T)
__global__ __launch_bounds__(256) void build_total(
    const float* __restrict__ cand, const float* __restrict__ nmap,
    const float* __restrict__ valmap, float* __restrict__ tot)
{
  __shared__ float rv[256];
  __shared__ float vsh;
  int tid = threadIdx.x;
  float bv = 0.f;
  for (int i = tid; i < HWO; i += 256) bv = fmaxf(bv, cand[i]);
  rv[tid] = bv; __syncthreads();
  for (int s = 128; s > 0; s >>= 1) { if (tid < s) rv[tid] = fmaxf(rv[tid], rv[tid + s]); __syncthreads(); }
  if (tid == 0) vsh = rv[0] * (float)T_STEPS;
  __syncthreads();
  float v = vsh;
  for (int i = tid; i < HWO; i += 256) {
    float n = nmap[i];
    tot[i] = (n > 0.f) ? n * (valmap[i] + v) : 0.f;
  }
}

// ---------------------------------------------------------------- iterative 5-winner KWTA
__global__ __launch_bounds__(256) void k_winners(
    float* __restrict__ tot, const int* __restrict__ winmap, float* __restrict__ wout)
{
  __shared__ float rv[256];
  __shared__ int   rk[256];
  __shared__ int   sf, sh, sw, svalid;
  int tid = threadIdx.x;
  for (int it = 0; it < 5; ++it) {
    float bv = -1.f; int bk = 0x7fffffff;
    for (int hw = tid; hw < HWO; hw += 256) {
      float v = tot[hw];
      int key = winmap[hw] * HWO + hw;        // flattened (F,H,W) index for tie-break
      if (v > bv || (v == bv && key < bk)) { bv = v; bk = key; }
    }
    rv[tid] = bv; rk[tid] = bk;
    __syncthreads();
    for (int s = 128; s > 0; s >>= 1) {
      if (tid < s) {
        float ov = rv[tid + s]; int ok = rk[tid + s];
        if (ov > rv[tid] || (ov == rv[tid] && ok < rk[tid])) { rv[tid] = ov; rk[tid] = ok; }
      }
      __syncthreads();
    }
    if (tid == 0) {
      float mv = rv[0]; int key = rk[0];
      int f = key / HWO, hw = key % HWO;
      int h = hw / WO, w = hw % WO;
      int valid = (mv > 0.f) ? 1 : 0;
      wout[it * 3 + 0] = valid ? (float)f : -1.f;
      wout[it * 3 + 1] = valid ? (float)h : -1.f;
      wout[it * 3 + 2] = valid ? (float)w : -1.f;
      sf = f; sh = h; sw = w; svalid = valid;
    }
    __syncthreads();
    if (svalid) {
      for (int hw = tid; hw < HWO; hw += 256) {
        int h = hw / WO, w = hw % WO;
        int dh = h - sh; if (dh < 0) dh = -dh;
        int dw = w - sw; if (dw < 0) dw = -dw;
        if (winmap[hw] == sf || (dh <= 3 && dw <= 3)) tot[hw] = 0.f;
      }
    }
    __syncthreads();
  }
}

// ---------------------------------------------------------------- launch
extern "C" void kernel_launch(void* const* d_in, const int* in_sizes, int n_in,
                              void* d_out, int out_size, void* d_ws, size_t ws_size,
                              hipStream_t stream)
{
  (void)in_sizes; (void)n_in; (void)out_size; (void)ws_size;
  const float* inp   = (const float*)d_in[0];   // (15,6,96,96)
  const float* wconv = (const float*)d_in[1];   // (512,6,5,5)
  const float* wrec  = (const float*)d_in[2];   // (512,512)
  // d_in[3] = max_layer (fixed == 2 in setup -> recurrent path always taken)

  float* out     = (float*)d_out;
  float* spk_out = out;                                       // T*NF*HWO
  float* pot_out = out + (size_t)T_STEPS * NF * HWO;          // T*NF*HWO
  float* win_out = out + 2 * (size_t)T_STEPS * NF * HWO;      // 15 floats

  char* ws = (char*)d_ws;
  size_t off = 0;
  _Float16* wconv_h = (_Float16*)(ws + off); off += (size_t)NF * KCONVP * 2;
  _Float16* wrec_h  = (_Float16*)(ws + off); off += (size_t)NF * NF * 2;
  _Float16* spk1h   = (_Float16*)(ws + off); off += (size_t)MROWS * NF * 2;
  int*   winmap = (int*)(ws + off);   off += (size_t)HWO * 4;
  float* nmap   = (float*)(ws + off); off += (size_t)HWO * 4;
  float* valmap = (float*)(ws + off); off += (size_t)HWO * 4;
  float* cand   = (float*)(ws + off); off += (size_t)HWO * 4;
  float* tot    = (float*)(ws + off); off += (size_t)HWO * 4;

  prep_weights<<<dim3((NF * NF + 255) / 256), 256, 0, stream>>>(wconv, wrec, wconv_h, wrec_h);
  conv_fire1  <<<dim3(6, HO, T_STEPS), 256, 0, stream>>>(inp, wconv_h, pot_out, spk1h);
  gemm_fire2  <<<dim3(MROWS / 16), 256, 0, stream>>>(spk1h, wrec_h, pot_out, spk_out);
  inhibit_stats<<<dim3(HWO), 256, 0, stream>>>(pot_out, winmap, nmap, valmap, cand);
  build_total <<<dim3(1), 256, 0, stream>>>(cand, nmap, valmap, tot);
  k_winners   <<<dim3(1), 256, 0, stream>>>(tot, winmap, win_out);
}
